// TGCNCell_60352880443527
// MI455X (gfx1250) — matrix operand order repaired and checked by reading, not compiled, not run on hardware
//
#include <hip/hip_runtime.h>
#include <hip/hip_bf16.h>

// ---------------- problem constants (match reference) ----------------
#define NN   8192          // num nodes
#define HH   64            // hidden size
#define KDIM 8192          // GEMM reduction depth (x columns)
#define NOUT 192           // fused output width: 128 (W1) + 64 (W2)
#define KSPLIT 2           // split-K factor for occupancy
#define BPAD 40            // LDS row pitch (ushorts): conflict-free + 16B aligned

typedef __attribute__((ext_vector_type(16))) __bf16 v16bf_t;
typedef __attribute__((ext_vector_type(8)))  float  v8f_t;

union FragA { unsigned int u[8]; v16bf_t v; };

#if defined(__has_builtin)
#if __has_builtin(__builtin_amdgcn_cvt_pk_bf16_f32)
#define HAVE_CVT_PK_BF16 1
#endif
#endif

__device__ __forceinline__ unsigned short f2bf(float f) {
    unsigned int u = __float_as_uint(f);
    unsigned int r = u + 0x7FFFu + ((u >> 16) & 1u);   // round to nearest even
    return (unsigned short)(r >> 16);
}

__device__ __forceinline__ unsigned int pack2bf(float a, float b) {
#ifdef HAVE_CVT_PK_BF16
    typedef __attribute__((ext_vector_type(2))) __bf16 v2bf_t;
    v2bf_t r = __builtin_amdgcn_cvt_pk_bf16_f32(a, b);  // lo = a, hi = b (RNE)
    return __builtin_bit_cast(unsigned int, r);
#else
    return (unsigned int)f2bf(a) | ((unsigned int)f2bf(b) << 16);
#endif
}

__device__ __forceinline__ float sigmoidf_(float x) {
    return 1.0f / (1.0f + __expf(-x));
}

// ---------------- degree / normalization ----------------
__global__ void k_deg_init(float* degf) {
    int i = blockIdx.x * blockDim.x + threadIdx.x;
    if (i < NN) degf[i] = 1.0f;                        // self loop
}

__global__ void k_deg_accum(const int* __restrict__ dst, float* degf, int E) {
    int e = blockIdx.x * blockDim.x + threadIdx.x;
    if (e < E) atomicAdd(&degf[dst[e]], 1.0f);
}

__global__ void k_dinv(const float* __restrict__ degf, float* dinv) {
    int i = blockIdx.x * blockDim.x + threadIdx.x;
    if (i < NN) {
        float d = degf[i];
        dinv[i] = (d > 0.0f) ? rsqrtf(d) : 0.0f;
    }
}

// ---------------- pack fused weight columns to bf16, column-major [192][8192] ----
__global__ void k_packW(const float* __restrict__ W1, const float* __restrict__ W2,
                        unsigned short* __restrict__ Wcb) {
    int idx = blockIdx.x * blockDim.x + threadIdx.x;   // NOUT * KDIM
    int n = idx >> 13;             // / KDIM
    int k = idx & (KDIM - 1);
    float v = (n < 128) ? W1[(size_t)k * 128 + n] : W2[(size_t)k * 64 + (n - 128)];
    Wcb[idx] = f2bf(v);            // Wcb[n*KDIM + k]
}

// ---------------- fused WMMA GEMM: P[y][8192,192] = x[:, Kslice] @ Wc[Kslice] ----
// Block tile 64(M) x 192(N), 8 waves (2x4), wave tile 32x48 (2x3 wmma tiles).
// Double-buffered LDS; prefetch for tile i+1 goes straight into LDS[buf^1]
// (no staging registers live across the barrier => no scratch spills).
__global__ __launch_bounds__(256, 1)
__attribute__((amdgpu_waves_per_eu(1)))
void k_gemm_wmma(const float* __restrict__ x,
                 const unsigned short* __restrict__ Wcb,
                 float* __restrict__ Pall) {
    __shared__ __align__(16) unsigned short a_sh[2][64][BPAD];   // [buf][row][k] bf16
    __shared__ __align__(16) unsigned short b_sh[2][192][BPAD];  // [buf][n][k]  bf16

    const int tid  = threadIdx.x;
    const int lane = tid & 31;
    const int w    = tid >> 5;        // 0..7
    const int wm   = w >> 2;          // 0..1  (M wave)
    const int wn   = w & 3;           // 0..3  (N wave)
    const int hi   = lane >> 4;       // lane half
    const int lr   = lane & 15;
    const int m0   = blockIdx.x * 64;
    const int kbase = blockIdx.y * (KDIM / KSPLIT);
    float* __restrict__ P = Pall + (size_t)blockIdx.y * NN * NOUT;

    // per-thread staging coordinates (fixed for whole kernel)
    const int ar0 = tid >> 3;                 // A row for i=0   (idx = tid)
    const int af0 = tid & 7;                  // A f4  for i=0
    const int ar1 = (tid + 256) >> 3;         // A row for i=1
    const int af1 = af0;                      // (tid+256)&7 == tid&7
    const int bn[3]  = { tid >> 2, (tid + 256) >> 2, (tid + 512) >> 2 };
    const int bpt    = tid & 3;               // (idx & 3) identical for all i

    v8f_t acc[2][3];
#pragma unroll
    for (int a = 0; a < 2; ++a)
#pragma unroll
        for (int b = 0; b < 3; ++b) acc[a][b] = (v8f_t)0.0f;

    const int NIT = (KDIM / KSPLIT) / 32;      // 128 iterations

    // ---- prologue: stage tile 0 into LDS[0] ----
    {
        const int kk = kbase;
        float4 a0 = *(const float4*)(x + (size_t)(m0 + ar0) * KDIM + kk + af0 * 4);
        float4 a1 = *(const float4*)(x + (size_t)(m0 + ar1) * KDIM + kk + af1 * 4);
        uint2 p0; p0.x = pack2bf(a0.x, a0.y); p0.y = pack2bf(a0.z, a0.w);
        uint2 p1; p1.x = pack2bf(a1.x, a1.y); p1.y = pack2bf(a1.z, a1.w);
        *(uint2*)(&a_sh[0][ar0][af0 * 4]) = p0;
        *(uint2*)(&a_sh[0][ar1][af1 * 4]) = p1;
#pragma unroll
        for (int i = 0; i < 3; ++i) {
            uint4 bv = *(const uint4*)(Wcb + (size_t)bn[i] * KDIM + kk + bpt * 8);
            *(uint4*)(&b_sh[0][bn[i]][bpt * 8]) = bv;
        }
    }
    __syncthreads();

    for (int it = 0; it < NIT; ++it) {
        const int buf = it & 1;

        // ---- fragment loads from LDS[buf] (only LDS reads this iteration) ----
        FragA afr[2], bfr[3];
#pragma unroll
        for (int sm = 0; sm < 2; ++sm) {
            const int row = wm * 32 + sm * 16 + lr;
            const unsigned short* ap = &a_sh[buf][row][0];
#pragma unroll
            for (int r = 0; r < 8; ++r) {
                // A: VGPR r holds K pair (2r or 16+2(r-4)), hi-lanes +8
                int k0 = ((r & 3) << 1) + ((r >> 2) << 4) + (hi << 3);
                afr[sm].u[r] = *(const unsigned int*)(ap + k0);
            }
        }
#pragma unroll
        for (int sn = 0; sn < 3; ++sn) {
            const int col = wn * 48 + sn * 16 + lr;
            const unsigned short* bp = &b_sh[buf][col][0];
#pragma unroll
            for (int r = 0; r < 8; ++r) {
                // B: VGPR r holds K pair (2r or 16+2(r-4)), hi-lanes +8
                int k0 = ((r & 3) << 1) + ((r >> 2) << 4) + (hi << 3);
                bfr[sn].u[r] = *(const unsigned int*)(bp + k0);
            }
        }

        // ---- prefetch next tile straight into LDS[buf^1] ----
        if (it + 1 < NIT) {
            const int kk = kbase + (it + 1) * 32;
            const int nb = buf ^ 1;
            float4 a0 = *(const float4*)(x + (size_t)(m0 + ar0) * KDIM + kk + af0 * 4);
            float4 a1 = *(const float4*)(x + (size_t)(m0 + ar1) * KDIM + kk + af1 * 4);
            uint4 bv0 = *(const uint4*)(Wcb + (size_t)bn[0] * KDIM + kk + bpt * 8);
            uint4 bv1 = *(const uint4*)(Wcb + (size_t)bn[1] * KDIM + kk + bpt * 8);
            uint4 bv2 = *(const uint4*)(Wcb + (size_t)bn[2] * KDIM + kk + bpt * 8);
            uint2 p0; p0.x = pack2bf(a0.x, a0.y); p0.y = pack2bf(a0.z, a0.w);
            uint2 p1; p1.x = pack2bf(a1.x, a1.y); p1.y = pack2bf(a1.z, a1.w);
            *(uint2*)(&a_sh[nb][ar0][af0 * 4]) = p0;
            *(uint2*)(&a_sh[nb][ar1][af1 * 4]) = p1;
            *(uint4*)(&b_sh[nb][bn[0]][bpt * 8]) = bv0;
            *(uint4*)(&b_sh[nb][bn[1]][bpt * 8]) = bv1;
            *(uint4*)(&b_sh[nb][bn[2]][bpt * 8]) = bv2;
        }

        // ---- WMMAs (depend only on the fragment ds_loads) ----
#pragma unroll
        for (int sm = 0; sm < 2; ++sm)
#pragma unroll
            for (int sn = 0; sn < 3; ++sn)
                acc[sm][sn] = __builtin_amdgcn_wmma_f32_16x16x32_bf16(
                    false, afr[sm].v, false, bfr[sn].v,
                    (short)0, acc[sm][sn], false, false);

        __syncthreads();
    }

    // --- store D: VGPR m -> row m (lanes 0-15) / 8+m (lanes 16-31), col = lane&15 ---
#pragma unroll
    for (int sm = 0; sm < 2; ++sm) {
        const int rbase = m0 + wm * 32 + sm * 16 + hi * 8;
#pragma unroll
        for (int sn = 0; sn < 3; ++sn) {
            const int col = wn * 48 + sn * 16 + lr;
#pragma unroll
            for (int m = 0; m < 8; ++m)
                P[(size_t)(rbase + m) * NOUT + col] = acc[sm][sn][m];
        }
    }
}

// ---------------- hidden tails: h = PA_slice + PB_slice + hrow @ Wtail ----------------
__global__ void k_tail(const float* __restrict__ Pall, int pofs,
                       const float* __restrict__ hsrc,
                       const float* __restrict__ Wtail, int fw,
                       float* __restrict__ out) {
    int idx = blockIdx.x * blockDim.x + threadIdx.x;   // NN * fw
    int i = idx / fw;
    int j = idx - i * fw;
    size_t po = (size_t)i * NOUT + pofs + j;
    float s = Pall[po] + Pall[po + (size_t)NN * NOUT];  // split-K partial sums
    const float* hrow = hsrc + (size_t)i * HH;
#pragma unroll 8
    for (int t = 0; t < HH; ++t) s += hrow[t] * Wtail[t * fw + j];
    out[idx] = s;
}

// ---------------- GCN aggregation ----------------
__global__ void k_agg_init(const float* __restrict__ h, const float* __restrict__ bias,
                           const float* __restrict__ dinv, float* __restrict__ out, int fw) {
    int idx = blockIdx.x * blockDim.x + threadIdx.x;   // NN * fw
    int i = idx / fw;
    int j = idx - i * fw;
    float di = dinv[i];
    out[idx] = bias[j] + h[idx] * di * di;             // self loop: norm = dinv^2
}

// 4 features per thread: FW/4 threads per edge
template <int FW>
__global__ void k_edge_scatter(const int* __restrict__ src, const int* __restrict__ dst,
                               const float* __restrict__ dinv,
                               const float* __restrict__ h, float* __restrict__ out,
                               int E) {
    const int QPE = FW / 4;                            // quads per edge
    long long gid = (long long)blockIdx.x * blockDim.x + threadIdx.x;
    if (gid >= (long long)E * QPE) return;
    int e = (int)(gid / QPE);
    int j = (int)(gid - (long long)e * QPE) * 4;
    int s = src[e], d = dst[e];
    float nm = dinv[s] * dinv[d];
    float4 v = *(const float4*)(h + (size_t)s * FW + j);
    float* o = out + (size_t)d * FW + j;
    atomicAdd(o + 0, v.x * nm);
    atomicAdd(o + 1, v.y * nm);
    atomicAdd(o + 2, v.z * nm);
    atomicAdd(o + 3, v.w * nm);
}

// ---------------- sigmoid + flatten-split (r,u) + r*h ----------------
__global__ void k_ru(const float* __restrict__ out1, const float* __restrict__ hidden,
                     float* __restrict__ rh, float* __restrict__ u) {
    int idx = blockIdx.x * blockDim.x + threadIdx.x;   // NN*HH
    if (idx >= NN * HH) return;
    float rv = sigmoidf_(out1[idx]);                   // first half of flat ru
    rh[idx]  = rv * hidden[idx];
    u[idx]   = sigmoidf_(out1[NN * HH + idx]);         // second half of flat ru
}

// ---------------- final GRU mix ----------------
__global__ void k_final(const float* __restrict__ out2, const float* __restrict__ u,
                        const float* __restrict__ hidden, float* __restrict__ out) {
    int idx = blockIdx.x * blockDim.x + threadIdx.x;
    if (idx >= NN * HH) return;
    float c = tanhf(out2[idx]);
    float uu = u[idx];
    out[idx] = uu * hidden[idx] + (1.0f - uu) * c;
}

extern "C" void kernel_launch(void* const* d_in, const int* in_sizes, int n_in,
                              void* d_out, int out_size, void* d_ws, size_t ws_size,
                              hipStream_t stream) {
    const float* x      = (const float*)d_in[0];
    const float* hidden = (const float*)d_in[1];
    const float* W1     = (const float*)d_in[2];   // [8256,128]
    const float* b1     = (const float*)d_in[3];
    const float* W2     = (const float*)d_in[4];   // [8256,64]
    const float* b2     = (const float*)d_in[5];
    const int*   ei     = (const int*)d_in[6];     // [2,E]
    const int E = in_sizes[6] / 2;
    const int* src = ei;
    const int* dst = ei + E;

    // workspace layout (floats)
    float* ws   = (float*)d_ws;
    float* Pall = ws;                               // KSPLIT * 8192*192
    float* h1   = Pall + (size_t)KSPLIT * NN * NOUT;// 8192*128
    float* out1 = h1   + (size_t)NN * 128;          // 8192*128
    float* rh   = out1 + (size_t)NN * 128;          // 8192*64
    float* u    = rh   + (size_t)NN * HH;           // 8192*64
    float* h2   = u    + (size_t)NN * HH;           // 8192*64
    float* out2 = h2   + (size_t)NN * HH;           // 8192*64
    float* degf = out2 + (size_t)NN * HH;           // 8192
    float* dinv = degf + NN;                        // 8192
    unsigned short* Wcb = (unsigned short*)(dinv + NN); // [192][8192] bf16

    const int T = 256;

    // normalization coefficients + weight pre-pack (independent small kernels)
    k_deg_init <<<NN / T, T, 0, stream>>>(degf);
    k_deg_accum<<<(E + T - 1) / T, T, 0, stream>>>(dst, degf, E);
    k_dinv     <<<NN / T, T, 0, stream>>>(degf, dinv);
    k_packW    <<<(NOUT * KDIM) / T, T, 0, stream>>>(W1, W2, Wcb);

    // fused x-GEMM for both convs (bf16 WMMA, fp32 accumulate, split-K=2)
    {
        dim3 grid(NN / 64, KSPLIT);
        k_gemm_wmma<<<grid, T, 0, stream>>>(x, Wcb, Pall);
    }

    // conv1: tail + aggregate + activation split
    k_tail     <<<(NN * 128) / T, T, 0, stream>>>(Pall, 0, hidden, W1 + (size_t)KDIM * 128, 128, h1);
    k_agg_init <<<(NN * 128) / T, T, 0, stream>>>(h1, b1, dinv, out1, 128);
    {
        long long tot = (long long)E * 32;          // FW/4 threads per edge
        k_edge_scatter<128><<<(unsigned)((tot + T - 1) / T), T, 0, stream>>>(src, dst, dinv, h1, out1, E);
    }
    k_ru       <<<(NN * HH) / T, T, 0, stream>>>(out1, hidden, rh, u);

    // conv2: tail + aggregate
    k_tail     <<<(NN * HH) / T, T, 0, stream>>>(Pall, 128, rh, W2 + (size_t)KDIM * 64, 64, h2);
    k_agg_init <<<(NN * HH) / T, T, 0, stream>>>(h2, b2, dinv, out2, 64);
    {
        long long tot = (long long)E * 16;
        k_edge_scatter<64><<<(unsigned)((tot + T - 1) / T), T, 0, stream>>>(src, dst, dinv, h2, out2, E);
    }

    // GRU mix
    k_final    <<<(NN * HH) / T, T, 0, stream>>>(out2, u, hidden, (float*)d_out);
}
